// Network_9929964388467
// MI455X (gfx1250) — matrix operand order repaired and checked by reading, not compiled, not run on hardware
//
#include <hip/hip_runtime.h>
#include <hip/hip_bf16.h>

// ---------------------------------------------------------------------------
// SLAYER SNN forward for MI455X (gfx1250, wave32, WMMA).
//
// Layer 1:  u1[b,n,t] = sum_p w1[n,p] * srm(t - tidx[b,p])   (WMMA f16 GEMM)
// spike1 :  serial scan over t with exact 2-state IIR refractory
// Layer 2:  u2[b,m,t] = sum_n w2[m,n] * s1[b,n,t]            (WMMA f16 GEMM)
// spike2 :  psp alpha-kernel as 2-state IIR + refractory IIR -> out [32,10,350]
// ---------------------------------------------------------------------------

typedef _Float16 v16h __attribute__((ext_vector_type(16)));
typedef _Float16 v8h  __attribute__((ext_vector_type(8)));
typedef float    v8f  __attribute__((ext_vector_type(8)));

#define T_BINS   350
#define T_PAD    352          // padded to 22 * 16
#define BATCH    32
#define NPIX     3072         // 3*32*32
#define NH       240          // hidden neurons
#define NH_PAD   256          // K padded for layer-2 WMMA
#define NO       10           // output neurons
#define NO_PAD   16
#define MT       5            // M-tiles (16 rows each) per wave in GEMM1
#define THETA    10.0f
#define R_REF    0.36787944117144233f   // e^-1   (tau_ref = 1)
#define R_SR     0.90483741803595957f   // e^-0.1 (tau_sr  = 10)
#define C_REF    54.365636569180902f    // SCALE_REF*THETA*e = 20e
#define C_SR     0.27182818284590452f   // e / tau_sr

// ---- workspace layout (bytes, 256-aligned) --------------------------------
static constexpr size_t alignUp(size_t x) { return (x + 255) & ~size_t(255); }
static constexpr size_t OFF_TIDX = 0;
static constexpr size_t SZ_TIDX  = alignUp(sizeof(int) * BATCH * NPIX);
static constexpr size_t OFF_W1H  = OFF_TIDX + SZ_TIDX;
static constexpr size_t SZ_W1H   = alignUp(sizeof(_Float16) * NH * NPIX);
static constexpr size_t OFF_W2H  = OFF_W1H + SZ_W1H;
static constexpr size_t SZ_W2H   = alignUp(sizeof(_Float16) * NO_PAD * NH_PAD);
static constexpr size_t OFF_SRM  = OFF_W2H + SZ_W2H;
static constexpr size_t SZ_SRM   = alignUp(sizeof(_Float16) * T_PAD);
static constexpr size_t OFF_U1   = OFF_SRM + SZ_SRM;
static constexpr size_t SZ_U1    = alignUp(sizeof(float) * BATCH * NH * T_PAD);
static constexpr size_t OFF_S1   = OFF_U1 + SZ_U1;
static constexpr size_t SZ_S1    = alignUp(sizeof(_Float16) * BATCH * NH_PAD * T_PAD);
static constexpr size_t OFF_U2   = OFF_S1 + SZ_S1;

// ---------------------------------------------------------------------------
// prep kernels
// ---------------------------------------------------------------------------
__global__ void k_tidx(const int* __restrict__ inp, const int* __restrict__ table,
                       int* __restrict__ tidx) {
  int i = blockIdx.x * blockDim.x + threadIdx.x;
  if (i < BATCH * NPIX) tidx[i] = table[inp[i]];
}

__global__ void k_w1h(const float* __restrict__ w1, _Float16* __restrict__ w1h) {
  int i = blockIdx.x * blockDim.x + threadIdx.x;
  if (i < NH * NPIX) w1h[i] = (_Float16)w1[i];
}

__global__ void k_w2h(const float* __restrict__ w2, _Float16* __restrict__ w2h) {
  int i = blockIdx.x * blockDim.x + threadIdx.x;
  if (i < NO_PAD * NH_PAD) {
    int m = i / NH_PAD, k = i % NH_PAD;
    w2h[i] = (m < NO && k < NH) ? (_Float16)w2[m * NH + k] : (_Float16)0.0f;
  }
}

__global__ void k_srm(_Float16* __restrict__ srm) {
  int i = blockIdx.x * blockDim.x + threadIdx.x;
  if (i < T_PAD) {
    float d = (float)i;
    srm[i] = (_Float16)((d * 0.1f) * __expf(1.0f - d * 0.1f));  // srm(0) == 0
  }
}

__global__ void k_zero16(uint4* __restrict__ p, int n16) {
  int i = blockIdx.x * blockDim.x + threadIdx.x;
  if (i < n16) p[i] = make_uint4(0u, 0u, 0u, 0u);
}

// ---------------------------------------------------------------------------
// Layer-1 GEMM+psp fused:  u1 = W1 x SRM_shift   via v_wmma_f32_16x16x32_f16.
// grid (22, 3, 32), block 32: one wave computes MT=5 M-tiles (80 rows) x 16
// time columns, sharing one B fragment (the LDS gather) per K chunk.
// A fragments for all 5 tiles are staged into distinct registers first so the
// 10 global_load_b128 issue as one clause and the 5 WMMAs run back-to-back.
// ---------------------------------------------------------------------------
__global__ void k_gemm1(const _Float16* __restrict__ w1h,
                        const int* __restrict__ tidx,
                        const _Float16* __restrict__ srm,
                        float* __restrict__ u1) {
  __shared__ int      tid_lds[NPIX];
  __shared__ _Float16 srm_lds[T_PAD];

  const int lane = threadIdx.x;
  const int b  = blockIdx.z;
  const int m0 = blockIdx.y * (16 * MT);
  const int n0 = blockIdx.x * 16;

  for (int i = lane; i < NPIX; i += 32) tid_lds[i] = tidx[b * NPIX + i];
  for (int i = lane; i < T_PAD; i += 32) srm_lds[i] = srm[i];
  __syncthreads();

  const int half  = lane >> 4;          // 0: lanes 0-15, 1: lanes 16-31
  const int abase = half * 16;          // A layout: lane half h holds K = 16h + i
  const int tn    = n0 + (lane & 15);   // output time column for B fragment

  const _Float16* w1row = w1h + (size_t)(m0 + (lane & 15)) * NPIX;
  v8f c[MT];
#pragma unroll
  for (int r = 0; r < MT; ++r) c[r] = (v8f){0.f,0.f,0.f,0.f,0.f,0.f,0.f,0.f};

  for (int kc = 0; kc < NPIX; kc += 32) {
    // Stage all MT A fragments (distinct regs -> one load clause, no per-WMMA
    // loadcnt stalls).
    v8h lo[MT], hi[MT];
#pragma unroll
    for (int r = 0; r < MT; ++r) {
      const _Float16* arow = w1row + (size_t)r * 16 * NPIX + kc;
      lo[r] = *(const v8h*)(arow + abase);
      hi[r] = *(const v8h*)(arow + abase + 8);
    }

    // B fragment (32x16 f16): lane half h holds K = kc + 16h + i, col = tn.
    // srm(0)==0, so clamping the shift to 0 is exact and branch-free.
    const int kb = kc + half * 16;
    const int* tp = &tid_lds[kb];       // 64-byte aligned: int4 LDS reads
    v16h bm;
#pragma unroll
    for (int q = 0; q < 4; ++q) {
      int4 tv = *(const int4*)(tp + 4 * q);
      int d0 = tn - tv.x; d0 = d0 > 0 ? d0 : 0;
      int d1 = tn - tv.y; d1 = d1 > 0 ? d1 : 0;
      int d2 = tn - tv.z; d2 = d2 > 0 ? d2 : 0;
      int d3 = tn - tv.w; d3 = d3 > 0 ? d3 : 0;
      bm[4 * q + 0] = srm_lds[d0];
      bm[4 * q + 1] = srm_lds[d1];
      bm[4 * q + 2] = srm_lds[d2];
      bm[4 * q + 3] = srm_lds[d3];
    }

    // 5 back-to-back WMMAs against the shared B fragment.
#pragma unroll
    for (int r = 0; r < MT; ++r) {
      v16h a;
#pragma unroll
      for (int j = 0; j < 8; ++j) { a[j] = lo[r][j]; a[j + 8] = hi[r][j]; }
      c[r] = __builtin_amdgcn_wmma_f32_16x16x32_f16(false, a, false, bm,
                                                    (short)0, c[r], false, false);
    }
  }

  // C/D layout: VGPR v -> row (tile_m0 + v + 8*half), col n0 + (lane&15)
#pragma unroll
  for (int r = 0; r < MT; ++r)
#pragma unroll
    for (int v = 0; v < 8; ++v)
      u1[((size_t)b * NH + m0 + 16 * r + v + 8 * half) * T_PAD + tn] = c[r][v];
}

// ---------------------------------------------------------------------------
// spike() layer 1: serial scan over t, exact alpha-IIR refractory state.
// one thread per (b, n) sequence; writes f16 spikes for the layer-2 WMMA.
// ---------------------------------------------------------------------------
__global__ void k_spike1(const float* __restrict__ u1, _Float16* __restrict__ s1h) {
  int i = blockIdx.x * blockDim.x + threadIdx.x;
  if (i >= BATCH * NH) return;
  int b = i / NH, n = i % NH;
  const float* urow = u1 + ((size_t)b * NH + n) * T_PAD;
  _Float16* srow = s1h + ((size_t)b * NH_PAD + n) * T_PAD;
  float A = 0.f, Bc = 0.f;
  for (int t = 0; t < T_BINS; ++t) {
    float v = urow[t] - C_REF * Bc;          // ref(t) = -20e * sum j*r^j
    float s = (v >= THETA) ? 1.0f : 0.0f;
    srow[t] = (_Float16)s;
    Bc = R_REF * (Bc + A);                   // B[t+1] = r*(B[t]+A[t])
    A  = R_REF * A + s;                      // A[t+1] = r*A[t] + s_t
  }
}

// ---------------------------------------------------------------------------
// Layer-2 GEMM: u2 = W2pad[16x256] x S1[256xT]  via WMMA. grid (22,1,32), blk 32
// ---------------------------------------------------------------------------
__global__ void k_gemm2(const _Float16* __restrict__ w2h,
                        const _Float16* __restrict__ s1h,
                        float* __restrict__ u2) {
  const int lane = threadIdx.x;
  const int b  = blockIdx.z;
  const int n0 = blockIdx.x * 16;
  const int m     = lane & 15;
  const int half  = lane >> 4;
  const int abase = half * 16;
  const int tn    = n0 + (lane & 15);

  const _Float16* w2row = w2h + (size_t)m * NH_PAD;
  const _Float16* sbat  = s1h + (size_t)b * NH_PAD * T_PAD;
  v8f c = {0.f, 0.f, 0.f, 0.f, 0.f, 0.f, 0.f, 0.f};

#pragma unroll
  for (int kc = 0; kc < NH_PAD; kc += 32) {
    v8h lo = *(const v8h*)(w2row + kc + abase);
    v8h hi = *(const v8h*)(w2row + kc + abase + 8);
    v16h a;
#pragma unroll
    for (int j = 0; j < 8; ++j) { a[j] = lo[j]; a[j + 8] = hi[j]; }

    const int kb = kc + half * 16;
    v16h bm;
#pragma unroll
    for (int i = 0; i < 16; ++i)
      bm[i] = sbat[(size_t)(kb + i) * T_PAD + tn];

    c = __builtin_amdgcn_wmma_f32_16x16x32_f16(false, a, false, bm,
                                               (short)0, c, false, false);
  }
#pragma unroll
  for (int v = 0; v < 8; ++v)
    u2[((size_t)b * NO_PAD + v + 8 * half) * T_PAD + tn] = c[v];
}

// ---------------------------------------------------------------------------
// psp (alpha IIR) + spike() for layer 2 -> final output [32,10,350] f32
// ---------------------------------------------------------------------------
__global__ void k_spike2(const float* __restrict__ u2, float* __restrict__ out) {
  int i = blockIdx.x * blockDim.x + threadIdx.x;
  if (i >= BATCH * NO) return;
  int b = i / NO, m = i % NO;
  const float* urow = u2 + ((size_t)b * NO_PAD + m) * T_PAD;
  float* orow = out + ((size_t)b * NO + m) * T_BINS;
  float P = 0.f, Q = 0.f, A = 0.f, Bc = 0.f;
  for (int t = 0; t < T_BINS; ++t) {
    Q = R_SR * (Q + P);                      // Q[t] = r2*(Q[t-1]+P[t-1])
    P = R_SR * P + urow[t];                  // P[t] = r2*P[t-1] + x[t]
    float y = C_SR * Q;                      // psp output
    float v = y - C_REF * Bc;
    float s = (v >= THETA) ? 1.0f : 0.0f;
    orow[t] = s;
    Bc = R_REF * (Bc + A);
    A  = R_REF * A + s;
  }
}

// ---------------------------------------------------------------------------
extern "C" void kernel_launch(void* const* d_in, const int* in_sizes, int n_in,
                              void* d_out, int out_size, void* d_ws, size_t ws_size,
                              hipStream_t stream) {
  (void)in_sizes; (void)n_in; (void)out_size; (void)ws_size;
  const int*   inp   = (const int*)d_in[0];
  const int*   table = (const int*)d_in[1];
  const float* w1    = (const float*)d_in[2];
  const float* w2    = (const float*)d_in[3];
  float*       out   = (float*)d_out;

  char* ws = (char*)d_ws;
  int*       tidx = (int*)(ws + OFF_TIDX);
  _Float16*  w1h  = (_Float16*)(ws + OFF_W1H);
  _Float16*  w2h  = (_Float16*)(ws + OFF_W2H);
  _Float16*  srm  = (_Float16*)(ws + OFF_SRM);
  float*     u1   = (float*)(ws + OFF_U1);
  _Float16*  s1h  = (_Float16*)(ws + OFF_S1);
  float*     u2   = (float*)(ws + OFF_U2);

  // prep
  k_tidx<<<(BATCH * NPIX + 255) / 256, 256, 0, stream>>>(inp, table, tidx);
  k_w1h <<<(NH * NPIX + 255) / 256, 256, 0, stream>>>(w1, w1h);
  k_w2h <<<(NO_PAD * NH_PAD + 255) / 256, 256, 0, stream>>>(w2, w2h);
  k_srm <<<(T_PAD + 255) / 256, 256, 0, stream>>>(srm);
  {
    int n16 = (int)(SZ_S1 / 16);
    k_zero16<<<(n16 + 255) / 256, 256, 0, stream>>>((uint4*)s1h, n16);
  }

  // layer 1: fused encode+psp+GEMM on WMMA, then serial spike scan
  k_gemm1<<<dim3(T_PAD / 16, NH / (16 * MT), BATCH), 32, 0, stream>>>(w1h, tidx, srm, u1);
  k_spike1<<<(BATCH * NH + 255) / 256, 256, 0, stream>>>(u1, s1h);

  // layer 2: WMMA GEMM, then psp-IIR + spike scan -> output
  k_gemm2<<<dim3(T_PAD / 16, 1, BATCH), 32, 0, stream>>>(w2h, s1h, u2);
  k_spike2<<<(BATCH * NO + 319) / 320, 320, 0, stream>>>(u2, out);
}
